// FlaxAttention_50672024158925
// MI455X (gfx1250) — matrix operand order repaired and checked by reading, not compiled, not run on hardware
//
#include <hip/hip_runtime.h>

typedef unsigned short u16;
typedef unsigned int   u32;
typedef __bf16 bf16x16 __attribute__((ext_vector_type(16)));
typedef float  f32x8   __attribute__((ext_vector_type(8)));
typedef int    v4i     __attribute__((vector_size(16)));

union FragAB {
    bf16x16 v;
    uint4   q[2];
};

__device__ inline u16 f32_to_bf16(float f) {
    u32 x = __float_as_uint(f);
    u32 r = x + 0x7FFFu + ((x >> 16) & 1u);   // round-to-nearest-even
    return (u16)(r >> 16);
}

__device__ inline f32x8 wmma_bf16(const FragAB& a, const FragAB& b, f32x8 c) {
    return __builtin_amdgcn_wmma_f32_16x16x32_bf16(
        false, a.v, false, b.v, (short)0, c, false, false);
}

// ---------------------------------------------------------------------------
// CDNA5 async global->LDS copy (ASYNCcnt path), with portable fallback.
// ---------------------------------------------------------------------------
#if defined(__has_builtin)
#  if __has_builtin(__builtin_amdgcn_global_load_async_to_lds_b128) && \
      __has_builtin(__builtin_amdgcn_s_wait_asynccnt)
#    define HAVE_ASYNC_LDS 1
#  endif
#endif
#ifndef HAVE_ASYNC_LDS
#  define HAVE_ASYNC_LDS 0
#endif

#if HAVE_ASYNC_LDS
__device__ inline void async_copy16(const u16* gptr, u16* lptr) {
    __builtin_amdgcn_global_load_async_to_lds_b128(
        (__attribute__((address_space(1))) v4i*)gptr,
        (__attribute__((address_space(3))) v4i*)lptr,
        0, 0);
}
__device__ inline void async_wait_all() { __builtin_amdgcn_s_wait_asynccnt(0); }
__device__ inline void async_wait_grp() { __builtin_amdgcn_s_wait_asynccnt(3); }
#else
__device__ inline void async_copy16(const u16* gptr, u16* lptr) {
    *(uint4*)lptr = *(const uint4*)gptr;
}
__device__ inline void async_wait_all() {}
__device__ inline void async_wait_grp() {}
#endif

// ---------------------------------------------------------------------------
// Kernel 1a: f32 -> bf16 cast (n4 = n/4 vector chunks)
// ---------------------------------------------------------------------------
__global__ __launch_bounds__(256) void cvt_f32_bf16(const float* __restrict__ src,
                                                    u16* __restrict__ dst, int n4) {
    int i = blockIdx.x * blockDim.x + threadIdx.x;
    if (i < n4) {
        float4 f = ((const float4*)src)[i];
        ushort4 o;
        o.x = f32_to_bf16(f.x);
        o.y = f32_to_bf16(f.y);
        o.z = f32_to_bf16(f.z);
        o.w = f32_to_bf16(f.w);
        ((ushort4*)dst)[i] = o;
    }
}

// ---------------------------------------------------------------------------
// Kernel 1b: f32 [K][N] -> bf16 transposed [N][K] (64x64 tiles via LDS)
// ---------------------------------------------------------------------------
__global__ __launch_bounds__(256) void cvt_transpose(const float* __restrict__ W,
                                                     u16* __restrict__ Wt,
                                                     int Kdim, int Ndim) {
    __shared__ u16 tile[64][72];   // padded rows to dodge bank conflicts
    const int tid = threadIdx.x;
    const int bk = blockIdx.y * 64;
    const int bn = blockIdx.x * 64;
#pragma unroll
    for (int i = 0; i < 4; ++i) {               // 64 rows x 16 float4 chunks
        int c = tid + i * 256;
        int r = c >> 4;                         // k row
        int col = (c & 15) * 4;                 // n col
        float4 f = *(const float4*)(W + (size_t)(bk + r) * Ndim + bn + col);
        tile[col + 0][r] = f32_to_bf16(f.x);
        tile[col + 1][r] = f32_to_bf16(f.y);
        tile[col + 2][r] = f32_to_bf16(f.z);
        tile[col + 3][r] = f32_to_bf16(f.w);
    }
    __syncthreads();
#pragma unroll
    for (int i = 0; i < 2; ++i) {               // 64 rows x 8 uint4 chunks
        int c = tid + i * 256;
        int r = c >> 3;                         // n row of Wt
        int cc = (c & 7) * 8;                   // k col
        *(uint4*)(Wt + (size_t)(bn + r) * Kdim + bk + cc) = *(uint4*)&tile[r][cc];
    }
}

// ---------------------------------------------------------------------------
// Kernel 2: bf16 GEMM  C[M,N] = A[M,K] @ Bt[N,K]^T, f32 accumulate.
// WG tile 128x64, 8 waves (4x2), each wave 32x32, K-step 32.
// Double-buffered LDS fed by async global->LDS copies.
// ---------------------------------------------------------------------------
#define GTM 128
#define GTN 64
#define GTK 32

template<bool F32OUT>
__global__ __launch_bounds__(256) void gemm_bf16(const u16* __restrict__ A,
                                                 const u16* __restrict__ Bt,
                                                 float* __restrict__ Cf,
                                                 u16* __restrict__ Cb,
                                                 int M, int N, int K) {
    __shared__ u16 Asl[2][GTM][GTK];   // 2 x 8 KB
    __shared__ u16 Bsl[2][GTN][GTK];   // 2 x 4 KB  (row = n, K contiguous)

    const int tid  = threadIdx.x;
    const int wave = tid >> 5;
    const int lane = tid & 31;
    const int g    = lane >> 4;
    const int lr   = lane & 15;

    const int m0 = blockIdx.y * GTM;
    const int n0 = blockIdx.x * GTN;
    const int wm = (wave & 3) * 32;
    const int wn = (wave >> 2) * 32;

    // fixed per-thread copy chunks (3 x 16B per K-step)
    const int ar0 = tid >> 2;                  // A rows 0..63
    const int ar1 = (tid >> 2) + 64;           // A rows 64..127
    const int ac  = (tid & 3) * 8;
    const int br  = tid >> 2;                  // B rows 0..63
    const int bc  = (tid & 3) * 8;

    auto issue = [&](int buf, int k0) {
        async_copy16(A  + (size_t)(m0 + ar0) * K + k0 + ac, &Asl[buf][ar0][ac]);
        async_copy16(A  + (size_t)(m0 + ar1) * K + k0 + ac, &Asl[buf][ar1][ac]);
        async_copy16(Bt + (size_t)(n0 + br)  * K + k0 + bc, &Bsl[buf][br][bc]);
    };

    f32x8 acc[2][2] = {};
    issue(0, 0);
    int kb = 0;
    for (int k0 = 0; k0 < K; k0 += GTK, kb ^= 1) {
        if (k0 + GTK < K) { issue(kb ^ 1, k0 + GTK); async_wait_grp(); }
        else              { async_wait_all(); }
        __syncthreads();

        FragAB af[2], bf[2];
#pragma unroll
        for (int i = 0; i < 2; ++i) {
            const u16* arow = &Asl[kb][wm + i * 16 + lr][0];
            af[i].q[0] = *(const uint4*)(arow + g * 8);
            af[i].q[1] = *(const uint4*)(arow + 16 + g * 8);
            const u16* brow = &Bsl[kb][wn + i * 16 + lr][0];
            bf[i].q[0] = *(const uint4*)(brow + g * 8);
            bf[i].q[1] = *(const uint4*)(brow + 16 + g * 8);
        }
#pragma unroll
        for (int i = 0; i < 2; ++i)
#pragma unroll
            for (int j = 0; j < 2; ++j)
                acc[i][j] = wmma_bf16(af[i], bf[j], acc[i][j]);
        __syncthreads();
    }

    // C layout: VGPR r, lane L -> M = r + 8*(L/16), N = L%16
#pragma unroll
    for (int i = 0; i < 2; ++i)
#pragma unroll
        for (int j = 0; j < 2; ++j)
#pragma unroll
            for (int r = 0; r < 8; ++r) {
                int row = m0 + wm + i * 16 + r + 8 * g;
                int col = n0 + wn + j * 16 + lr;
                size_t idx = (size_t)row * N + col;
                float val = acc[i][j][r];
                if (F32OUT) Cf[idx] = val;
                else        Cb[idx] = f32_to_bf16(val);
            }
}

// ---------------------------------------------------------------------------
// Kernel 3: flash attention. One WG per (128-query block, head, batch).
// 8 waves, each owns 16 query rows; loops over 16 key blocks of 128.
// ---------------------------------------------------------------------------
__global__ __launch_bounds__(256) void attn_kernel(const u16* __restrict__ Qb,
                                                   const u16* __restrict__ Kb,
                                                   const u16* __restrict__ Vb,
                                                   u16* __restrict__ Ob) {
    __shared__ u16 QPs[8][16][128];   // Q tile [128][64], later 8 P slabs [16][128]
    __shared__ u16 Ks[128][64];       // t rows, d contiguous
    __shared__ u16 Vts[64][128];      // transposed: d rows, t contiguous

    const int tid  = threadIdx.x;
    const int wave = tid >> 5;
    const int lane = tid & 31;
    const int g    = lane >> 4;
    const int lr   = lane & 15;

    const int f0 = blockIdx.x * 128;
    const int h  = blockIdx.y;
    const int b  = blockIdx.z;
    const size_t brow = (size_t)b * 2048;
    const float scale = 0.125f;       // 1/sqrt(64)

    u16* Qflat = &QPs[0][0][0];
    u16* Pw    = &QPs[wave][0][0];

    // ---- load Q tile (async direct to LDS) ----
#pragma unroll
    for (int i = 0; i < 4; ++i) {
        int c = tid + i * 256;
        int row = c >> 3, col = (c & 7) * 8;
        async_copy16(Qb + (brow + f0 + row) * 1024 + h * 64 + col,
                     Qflat + row * 64 + col);
    }
    async_wait_all();
    __syncthreads();

    // ---- cache Q A-fragments (2 K-chunks of 32) ----
    FragAB qf[2];
    {
        const u16* qrow = Qflat + (wave * 16 + lr) * 64;
#pragma unroll
        for (int kc = 0; kc < 2; ++kc) {
            qf[kc].q[0] = *(const uint4*)(qrow + kc * 32 + g * 8);
            qf[kc].q[1] = *(const uint4*)(qrow + kc * 32 + 16 + g * 8);
        }
    }

    f32x8 o[4] = {};
    float m[8], l[8];
#pragma unroll
    for (int r = 0; r < 8; ++r) { m[r] = -INFINITY; l[r] = 0.0f; }

    for (int t0 = 0; t0 < 2048; t0 += 128) {
        __syncthreads();   // all waves done reading prior K/V (and Q frags)

        // K tile: async direct to LDS
#pragma unroll
        for (int i = 0; i < 4; ++i) {
            int c = tid + i * 256;
            int row = c >> 3, col = (c & 7) * 8;
            async_copy16(Kb + (brow + t0 + row) * 1024 + h * 64 + col,
                         &Ks[row][col]);
        }
        // V tile: batch 4 global loads into regs, then scatter transposed
        uint4 vr[4];
#pragma unroll
        for (int i = 0; i < 4; ++i) {
            int c = tid + i * 256;
            int row = c >> 3, col = (c & 7) * 8;
            vr[i] = *(const uint4*)(Vb + (brow + t0 + row) * 1024 + h * 64 + col);
        }
#pragma unroll
        for (int i = 0; i < 4; ++i) {
            int c = tid + i * 256;
            int row = c >> 3, col = (c & 7) * 8;
            const u16* e = (const u16*)&vr[i];
#pragma unroll
            for (int j = 0; j < 8; ++j) Vts[col + j][row] = e[j];
        }
        async_wait_all();
        __syncthreads();

        // ---- S = Q @ K^T : [16 f][128 t] in 8 column tiles ----
        f32x8 s[8];
#pragma unroll
        for (int nt = 0; nt < 8; ++nt) {
            f32x8 sc = {};
#pragma unroll
            for (int kc = 0; kc < 2; ++kc) {
                FragAB kf;  // B frag: N = t (lane), K = d = kc*32 + 8g + e (e<8), kc*32 + 16 + 8g + (e-8) (e>=8)
                const u16* krow = &Ks[nt * 16 + lr][0];
                kf.q[0] = *(const uint4*)(krow + kc * 32 + g * 8);
                kf.q[1] = *(const uint4*)(krow + kc * 32 + 16 + g * 8);
                sc = wmma_bf16(qf[kc], kf, sc);
            }
            s[nt] = sc;
        }

        // ---- online softmax ----
        float mnew[8];
#pragma unroll
        for (int r = 0; r < 8; ++r) {
            float lm = -INFINITY;
#pragma unroll
            for (int nt = 0; nt < 8; ++nt) {
                s[nt][r] *= scale;
                lm = fmaxf(lm, s[nt][r]);
            }
#pragma unroll
            for (int w = 1; w < 16; w <<= 1)
                lm = fmaxf(lm, __shfl_xor(lm, w, 32));
            mnew[r] = fmaxf(m[r], lm);
        }
#pragma unroll
        for (int r = 0; r < 8; ++r) {
            float alpha = __expf(m[r] - mnew[r]);
            m[r] = mnew[r];
            l[r] *= alpha;
#pragma unroll
            for (int j = 0; j < 4; ++j) o[j][r] *= alpha;
        }

        // ---- P = exp(S - m) -> bf16 into wave-private LDS slab ----
#pragma unroll
        for (int nt = 0; nt < 8; ++nt)
#pragma unroll
            for (int r = 0; r < 8; ++r) {
                float p = __expf(s[nt][r] - m[r]);
                s[nt][r] = p;
                Pw[(r + 8 * g) * 128 + nt * 16 + lr] = f32_to_bf16(p);
            }
#pragma unroll
        for (int r = 0; r < 8; ++r) {
            float rs = 0.0f;
#pragma unroll
            for (int nt = 0; nt < 8; ++nt) rs += s[nt][r];
#pragma unroll
            for (int w = 1; w < 16; w <<= 1) rs += __shfl_xor(rs, w, 32);
            l[r] += rs;
        }

        // ---- O += P @ V ----
#pragma unroll
        for (int kc = 0; kc < 4; ++kc) {
            FragAB pf;
            const u16* prow = Pw + lr * 128;
            pf.q[0] = *(const uint4*)(prow + kc * 32 + g * 8);
            pf.q[1] = *(const uint4*)(prow + kc * 32 + 16 + g * 8);
#pragma unroll
            for (int j = 0; j < 4; ++j) {
                FragAB vf;  // B frag: N = d (lane), K = t = kc*32 + 8g + e (e<8), kc*32 + 16 + 8g + (e-8) (e>=8)
                const u16* vrow = &Vts[j * 16 + lr][0];
                vf.q[0] = *(const uint4*)(vrow + kc * 32 + g * 8);
                vf.q[1] = *(const uint4*)(vrow + kc * 32 + 16 + g * 8);
                o[j] = wmma_bf16(pf, vf, o[j]);
            }
        }
    }

    // ---- normalize and store bf16 output [B*S, H*Dh] ----
#pragma unroll
    for (int r = 0; r < 8; ++r) {
        float inv = 1.0f / l[r];
        size_t row = brow + f0 + wave * 16 + r + 8 * g;
#pragma unroll
        for (int j = 0; j < 4; ++j)
            Ob[row * 1024 + h * 64 + j * 16 + lr] = f32_to_bf16(o[j][r] * inv);
    }
}

// ---------------------------------------------------------------------------
// Host-side orchestration
// ---------------------------------------------------------------------------
extern "C" void kernel_launch(void* const* d_in, const int* in_sizes, int n_in,
                              void* d_out, int out_size, void* d_ws, size_t ws_size,
                              hipStream_t stream) {
    const float* H  = (const float*)d_in[0];
    const float* Wq = (const float*)d_in[1];
    const float* Wk = (const float*)d_in[2];
    const float* Wv = (const float*)d_in[3];
    const float* Wo = (const float*)d_in[4];
    float* out = (float*)d_out;

    const int M = 4 * 2048;   // 8192 rows
    const int E = 1024;
    const size_t nH = (size_t)M * E;
    const size_t nW = (size_t)E * E;

    u16* ws  = (u16*)d_ws;
    u16* hb  = ws;                 size_t off = nH;
    u16* wqt = ws + off;           off += nW;   // transposed bf16 weights [N][K]
    u16* wkt = ws + off;           off += nW;
    u16* wvt = ws + off;           off += nW;
    u16* wot = ws + off;           off += nW;
    u16* Qb  = ws + off;           off += nH;
    u16* Kb  = ws + off;           off += nH;
    u16* Vb  = ws + off;           off += nH;
    u16* AOb = ws + off;           off += nH;
    (void)ws_size; (void)in_sizes; (void)n_in; (void)out_size;

    // casts (+ weight transpose)
    cvt_f32_bf16<<<(int)(nH / 4 / 256), 256, 0, stream>>>(H, hb, (int)(nH / 4));
    dim3 tg(E / 64, E / 64);   // (16,16)
    cvt_transpose<<<tg, 256, 0, stream>>>(Wq, wqt, E, E);
    cvt_transpose<<<tg, 256, 0, stream>>>(Wk, wkt, E, E);
    cvt_transpose<<<tg, 256, 0, stream>>>(Wv, wvt, E, E);
    cvt_transpose<<<tg, 256, 0, stream>>>(Wo, wot, E, E);

    // QKV projections (bf16 out)
    dim3 gg(E / GTN, M / GTM);    // (16, 64)
    gemm_bf16<false><<<gg, 256, 0, stream>>>(hb, wqt, nullptr, Qb, M, E, E);
    gemm_bf16<false><<<gg, 256, 0, stream>>>(hb, wkt, nullptr, Kb, M, E, E);
    gemm_bf16<false><<<gg, 256, 0, stream>>>(hb, wvt, nullptr, Vb, M, E, E);

    // attention: grid (S/128, H, B)
    attn_kernel<<<dim3(16, 16, 4), 256, 0, stream>>>(Qb, Kb, Vb, AOb);

    // output projection (f32 out)
    gemm_bf16<true><<<gg, 256, 0, stream>>>(AOb, wot, out, nullptr, M, E, E);
}
